// CrossAttentionLayerTopK_22883585753593
// MI455X (gfx1250) — matrix operand order repaired and checked by reading, not compiled, not run on hardware
//
#include <hip/hip_runtime.h>
#include <math.h>

// ---------------- problem constants ----------------
#define BSZ 2
#define CCH 64
#define NQ  400
#define NP  416          // keys padded to 13 chunks of 32
#define LQ  12
#define DM  128
#define NH  8
#define HD  16
#define FFD 1024
#define BL  (BSZ*LQ)     // 24 (b,l) slabs
#define NT  (NQ/16)      // 25 query tiles
#define KT  (NP/32)      // 13 K-chunks for P@V

typedef __attribute__((ext_vector_type(16))) _Float16 v16h;
typedef __attribute__((ext_vector_type(8)))  _Float16 v8h;
typedef __attribute__((ext_vector_type(8)))  float    v8f;

// packed-weight offsets (f16 elements), B-tile layout: tile(ct,kt) = 512 f16,
// element W[(32*kt + 16*(lane>>4) + e)*N + 16*ct + (lane&15)] at tile[lane*16+e]
#define PW_Q1 0
#define PW_Q2 8192
#define PW_K1 16384
#define PW_K2 24576
#define PW_V  32768
#define PW_O  40960       // N=64,K=128 : 4*4*512   = 8192
#define PW_F1 49152       // N=1024,K=64: 64*2*512  = 65536
#define PW_F2 114688      // N=64,K=1024: 4*32*512  = 65536
#define PW_TOT 180224

// ---------------- WMMA helpers ----------------
__device__ __forceinline__ v8f wmma_f16(v16h a, v16h b, v8f c) {
  return __builtin_amdgcn_wmma_f32_16x16x32_f16(
      false, a, false, b, (short)0, c, false, false);
}

// A operand, 16-bit 16x32: lane holds 16 contiguous f16
// at K = 16*half+[0..15]. `row` = this lane's M-row.
__device__ __forceinline__ v16h lda_row(const _Float16* row, int half, int koff) {
  const v8h lo = *(const v8h*)(row + koff + 16*half);
  const v8h hi = *(const v8h*)(row + koff + 16*half + 8);
  v16h r;
#pragma unroll
  for (int j = 0; j < 8; ++j) { r[j] = lo[j]; r[j+8] = hi[j]; }
  return r;
}

// B operand from a row-major 32-wide row (keys): 16 contiguous f16 at 16*half.
__device__ __forceinline__ v16h ldb_row(const _Float16* row, int half) {
  const v8h lo = *(const v8h*)(row + 16*half);
  const v8h hi = *(const v8h*)(row + 16*half + 8);
  v16h r;
#pragma unroll
  for (int j = 0; j < 8; ++j) { r[j] = lo[j]; r[j+8] = hi[j]; }
  return r;
}

// B operand from pre-packed tile: lane's 16 f16 contiguous at tile+lane*16.
__device__ __forceinline__ v16h ldb_packed(const _Float16* tile, int lane) {
  const v8h lo = *(const v8h*)(tile + lane * 16);
  const v8h hi = *(const v8h*)(tile + lane * 16 + 8);
  v16h r;
#pragma unroll
  for (int j = 0; j < 8; ++j) { r[j] = lo[j]; r[j+8] = hi[j]; }
  return r;
}

__device__ __forceinline__ float wsum(float v) {
#pragma unroll
  for (int m = 1; m < 32; m <<= 1) v += __shfl_xor(v, m);
  return v;
}
__device__ __forceinline__ int wisum(int v) {
#pragma unroll
  for (int m = 1; m < 32; m <<= 1) v += __shfl_xor(v, m);
  return v;
}
__device__ __forceinline__ float wmax(float v) {
#pragma unroll
  for (int m = 1; m < 32; m <<= 1) v = fmaxf(v, __shfl_xor(v, m));
  return v;
}
__device__ __forceinline__ float wmin(float v) {
#pragma unroll
  for (int m = 1; m < 32; m <<= 1) v = fminf(v, __shfl_xor(v, m));
  return v;
}

// kth-largest threshold over a 400(+pad=-inf) row, one wave, binary search.
__device__ __forceinline__ float kth_thresh(const float* vals, float lo, float hi, int k) {
#pragma unroll 1
  for (int it = 0; it < 30; ++it) {
    float mid = 0.5f * (lo + hi);
    int cnt = 0;
#pragma unroll
    for (int i = 0; i < 13; ++i) cnt += (vals[i] >= mid) ? 1 : 0;
    cnt = wisum(cnt);
    if (cnt >= k) lo = mid; else hi = mid;
  }
  return lo;
}

// ---------------- kernel 0a: zero f16 staging (V pads must be 0) ------------
__global__ void zero_f16_kernel(_Float16* p, long n) {
  long i = (long)blockIdx.x * blockDim.x + threadIdx.x;
  if (i < n) p[i] = (_Float16)0.0f;
}

// ---------------- kernel 0b: pack a weight matrix into B-tile f16 layout ----
// W is [K x N] row-major f32. dst tile(ct,kt) at (ct*(K/32)+kt)*512.
__global__ void pack_w_kernel(const float* __restrict__ W, _Float16* __restrict__ dst,
                              int N, int K) {
  int idx = blockIdx.x * blockDim.x + threadIdx.x;
  if (idx >= N * K) return;
  int tile = idx >> 9, within = idx & 511;
  int lane = within >> 4, e = within & 15;
  int ktiles = K >> 5;
  int ct = tile / ktiles, kt = tile % ktiles;
  int row = 32 * kt + 16 * (lane >> 4) + e;
  int col = 16 * ct + (lane & 15);
  dst[idx] = (_Float16)W[(size_t)row * N + col];
}

// ---------------- kernel 1: QKV projections + l2norm ------------------------
// grid (25, 24), 128 threads (4 waves), per-wave uniform control flow.
__global__ void proj_kernel(const float* __restrict__ x,
                            const float* __restrict__ xa1,
                            const float* __restrict__ xa2,
                            const _Float16* __restrict__ pw,
                            const float* __restrict__ bQ1, const float* __restrict__ bQ2,
                            const float* __restrict__ bK1, const float* __restrict__ bK2,
                            const float* __restrict__ bV,
                            _Float16* __restrict__ qc, _Float16* __restrict__ kc,
                            _Float16* __restrict__ vbp) {
  const int nt = blockIdx.x, bl = blockIdx.y;
  const int b = bl / LQ, l = bl % LQ, n0 = nt * 16;
  __shared__ _Float16 XT[16][64], A1[16][64], A2[16][64];

  const int tid = threadIdx.x;
  for (int idx = tid; idx < 16 * 64; idx += blockDim.x) {
    int tok = idx >> 6, ch = idx & 63;
    size_t gi = ((size_t)(b * CCH + ch) * NQ + (n0 + tok)) * LQ + l;
    XT[tok][ch] = (_Float16)x[gi];
    A1[tok][ch] = (_Float16)xa1[gi];
    A2[tok][ch] = (_Float16)xa2[gi];
  }
  __syncthreads();

  const int wave = tid >> 5, lane = tid & 31;
  const int ln = lane & 15, half = lane >> 4;

  // ---- Q1/Q2/K1/K2: combos 0..31 (mat = combo>>3, head tile = combo&7) ----
#pragma unroll 1
  for (int combo = wave; combo < 32; combo += 4) {
    const int mat = combo >> 3, ct = combo & 7, d0 = ct * 16;
    const _Float16 (*src)[64] = (mat == 0) ? A1 : ((mat == 1) ? A2 : XT);
    const _Float16* pmat = pw + ((mat == 0) ? PW_Q1 : (mat == 1) ? PW_Q2
                                : (mat == 2) ? PW_K1 : PW_K2);
    const float* bb = (mat == 0) ? bQ1 : (mat == 1) ? bQ2
                    : (mat == 2) ? bK1 : bK2;
    v8f acc = {};
#pragma unroll
    for (int kt = 0; kt < 2; ++kt) {
      v16h a = lda_row(&src[ln][0], half, 32 * kt);
      v16h bm = ldb_packed(pmat + (size_t)(ct * 2 + kt) * 512, lane);
      acc = wmma_f16(a, bm, acc);
    }
    const float bias = bb[d0 + ln];
#pragma unroll
    for (int r = 0; r < 8; ++r) acc[r] += bias;
    // per-(token, head) l2norm across the 16-lane half-group
#pragma unroll
    for (int r = 0; r < 8; ++r) {
      float ss = acc[r] * acc[r];
      ss += __shfl_xor(ss, 1); ss += __shfl_xor(ss, 2);
      ss += __shfl_xor(ss, 4); ss += __shfl_xor(ss, 8);
      acc[r] *= 1.0f / fmaxf(sqrtf(ss), 1e-12f);
    }
    _Float16* dst = (mat <= 1) ? qc : kc;
    const int coff = (mat == 1 || mat == 3) ? 16 : 0;  // q2/k2 -> cols 16..31
    const size_t base = ((size_t)(bl * NH + ct) * NP) * 32;
#pragma unroll
    for (int r = 0; r < 8; ++r) {
      int row = r + 8 * half;
      dst[base + (size_t)(n0 + row) * 32 + coff + ln] = (_Float16)acc[r];
    }
  }

  // ---- V: head tiles 0..7, stored directly in packed B-tile layout --------
#pragma unroll 1
  for (int ct = wave; ct < 8; ct += 4) {
    v8f acc = {};
#pragma unroll
    for (int kt = 0; kt < 2; ++kt) {
      v16h a = lda_row(&XT[ln][0], half, 32 * kt);
      v16h bm = ldb_packed(pw + PW_V + (size_t)(ct * 2 + kt) * 512, lane);
      acc = wmma_f16(a, bm, acc);
    }
    const float bias = bV[ct * 16 + ln];
    const size_t hbase = ((size_t)(bl * NH + ct)) * (KT * 512);
#pragma unroll
    for (int r = 0; r < 8; ++r) {
      int tok = n0 + r + 8 * half;               // K-dim of the P@V B operand
      int bkt = tok >> 5, kk = tok & 31;
      int lane_b = (kk >> 4) * 16 + ln;          // (half_b, col=d)
      vbp[hbase + (size_t)bkt * 512 + lane_b * 16 + (kk & 15)] =
          (_Float16)(acc[r] + bias);
    }
  }
}

// ------- kernel 2: scores (one WMMA = q1k1+q2k2), top-k, softmax, P@V -------
// grid (25, 24, 8), 32 threads (one wave).
__global__ void attn_kernel(const _Float16* __restrict__ qc,
                            const _Float16* __restrict__ kc,
                            const _Float16* __restrict__ vbp,
                            const float* __restrict__ temperature,
                            const float* __restrict__ alphas,
                            float* __restrict__ out_attn,
                            float* __restrict__ out_p2,
                            _Float16* __restrict__ aoh) {
  const int nt = blockIdx.x, bl = blockIdx.y, h = blockIdx.z;
  const int b = bl / LQ, l = bl % LQ, n0 = nt * 16;
  const int lane = threadIdx.x, ln = lane & 15, half = lane >> 4;
  const float temp = temperature[h];

  __shared__ float    S[16][NP];
  __shared__ _Float16 P1[16][NP];
  __shared__ _Float16 P2[16][NP];

  const size_t qkb = ((size_t)(bl * NH + h) * NP) * 32;
  const v16h aq = lda_row(qc + qkb + (size_t)(n0 + ln) * 32, half, 0);

  // scores for 25 key tiles (cols 0..399)
#pragma unroll 1
  for (int mt = 0; mt < 25; ++mt) {
    if (mt + 1 < 25)
      __builtin_prefetch(kc + qkb + (size_t)((mt + 1) * 16 + ln) * 32, 0, 1);
    v16h bk = ldb_row(kc + qkb + (size_t)(mt * 16 + ln) * 32, half);
    v8f s = {};
    s = wmma_f16(aq, bk, s);
#pragma unroll
    for (int r = 0; r < 8; ++r)
      S[r + 8 * half][mt * 16 + ln] = s[r] * temp;
  }
  // pad cols 400..415 = -inf
  for (int idx = lane; idx < 16 * 16; idx += 32)
    S[idx >> 4][NQ + (idx & 15)] = -INFINITY;
  __syncthreads();

  const size_t abase = (((size_t)(b * NH + h) * LQ + l) * NQ + n0) * (size_t)NQ;
  // stream raw (scaled) scores out
#pragma unroll 1
  for (int q = 0; q < 16; ++q)
    for (int c = lane; c < NQ; c += 32)
      out_attn[abase + (size_t)q * NQ + c] = S[q][c];

  // per-row top-k thresholds + two softmaxes
#pragma unroll 1
  for (int q = 0; q < 16; ++q) {
    float vals[13];
    float vmx = -3.4e38f, vmn = 3.4e38f;
#pragma unroll
    for (int i = 0; i < 13; ++i) {
      int c = lane + 32 * i;
      float v = S[q][c];
      vals[i] = v;
      if (c < NQ) { vmx = fmaxf(vmx, v); vmn = fminf(vmn, v); }
    }
    vmx = wmax(vmx); vmn = wmin(vmn);
    const float t1 = kth_thresh(vals, vmn, vmx, 200);  // N/2
    const float t2 = kth_thresh(vals, vmn, vmx, 100);  // N/4
    float s1 = 0.f, s2 = 0.f;
#pragma unroll
    for (int i = 0; i < 13; ++i) {
      float e = __expf(vals[i] - vmx);
      if (vals[i] >= t1) s1 += e;
      if (vals[i] >= t2) s2 += e;
    }
    s1 = wsum(s1); s2 = wsum(s2);
    const float i1 = 1.0f / s1, i2 = 1.0f / s2;
#pragma unroll
    for (int i = 0; i < 13; ++i) {
      int c = lane + 32 * i;
      float e = __expf(vals[i] - vmx);
      float p1 = (vals[i] >= t1) ? e * i1 : 0.0f;
      float p2 = (vals[i] >= t2) ? e * i2 : 0.0f;
      P1[q][c] = (_Float16)p1;
      P2[q][c] = (_Float16)p2;
      if (c < NQ) out_p2[abase + (size_t)q * NQ + c] = p2;
    }
  }
  __syncthreads();

  // O1 = P1 @ V, O2 = P2 @ V  (13 chunks of K=32, packed V, zero-padded)
  v8f o1 = {}, o2 = {};
  const _Float16* vt = vbp + ((size_t)(bl * NH + h)) * (KT * 512);
#pragma unroll 1
  for (int kt = 0; kt < KT; ++kt) {
    if (kt + 1 < KT)
      __builtin_prefetch(vt + (size_t)(kt + 1) * 512 + lane * 16, 0, 1);
    v16h a1 = lda_row(&P1[ln][0], half, 32 * kt);
    v16h a2 = lda_row(&P2[ln][0], half, 32 * kt);
    v16h bv = ldb_packed(vt + (size_t)kt * 512, lane);
    o1 = wmma_f16(a1, bv, o1);
    o2 = wmma_f16(a2, bv, o2);
  }
  const float a0 = alphas[0], a1s = alphas[1];
  const float mx = fmaxf(a0, a1s);
  const float e0 = __expf(a0 - mx), e1 = __expf(a1s - mx);
  const float al0 = e0 / (e0 + e1), al1 = e1 / (e0 + e1);
#pragma unroll
  for (int r = 0; r < 8; ++r) {
    int row = r + 8 * half;
    aoh[((size_t)bl * NQ + n0 + row) * DM + h * 16 + ln] =
        (_Float16)(al0 * o1[r] + al1 * o2[r]);
  }
}

// -------- kernel 3: Wo + residual + LN1 + FFN + residual + LN2 + store ------
// grid (25, 24), 32 threads (one wave).
__global__ void epi_kernel(const float* __restrict__ x,
                           const _Float16* __restrict__ aoh,
                           const _Float16* __restrict__ pw,
                           const float* __restrict__ bo,
                           const float* __restrict__ bf1,
                           const float* __restrict__ bf2,
                           const float* __restrict__ g1, const float* __restrict__ be1,
                           const float* __restrict__ g2, const float* __restrict__ be2,
                           float* __restrict__ out) {
  const int nt = blockIdx.x, bl = blockIdx.y;
  const int b = bl / LQ, l = bl % LQ, n0 = nt * 16;
  const int lane = threadIdx.x, ln = lane & 15, half = lane >> 4;

  __shared__ float    XT[16][64];
  __shared__ float    R1[16][64];     // pre-LN1 sum, then LN1 output (f32)
  __shared__ _Float16 L1H[16][64];    // LN1 output (f16 A operand)
  __shared__ _Float16 FFH[16][FFD];   // relu hidden (f16 A operand)
  __shared__ float    R2[16][64];

  for (int idx = lane; idx < 16 * 64; idx += 32) {
    int tok = idx >> 6, ch = idx & 63;
    XT[tok][ch] = x[((size_t)(b * CCH + ch) * NQ + n0 + tok) * LQ + l];
  }

  // y = ao @ Wo + bo  (K=128 -> 4 chunks, N=64 -> 4 col tiles)
  const _Float16* arow = aoh + ((size_t)bl * NQ + n0 + ln) * DM;
#pragma unroll 1
  for (int c4 = 0; c4 < 4; ++c4) {
    v8f acc = {};
#pragma unroll
    for (int kt = 0; kt < 4; ++kt) {
      v16h a = lda_row(arow, half, 32 * kt);
      v16h bm = ldb_packed(pw + PW_O + (size_t)(c4 * 4 + kt) * 512, lane);
      acc = wmma_f16(a, bm, acc);
    }
    const float bias = bo[c4 * 16 + ln];
#pragma unroll
    for (int r = 0; r < 8; ++r)
      R1[r + 8 * half][c4 * 16 + ln] = acc[r] + bias;
  }
  __syncthreads();

  // LN1 over (xt + y) across 64 channels
#pragma unroll 1
  for (int q = 0; q < 16; ++q) {
    float x0 = R1[q][lane]      + XT[q][lane];
    float x1 = R1[q][lane + 32] + XT[q][lane + 32];
    float mean = wsum(x0 + x1) * (1.0f / 64.0f);
    float var  = wsum(x0 * x0 + x1 * x1) * (1.0f / 64.0f) - mean * mean;
    float rs = rsqrtf(var + 1e-5f);
    float y0 = (x0 - mean) * rs * g1[lane]      + be1[lane];
    float y1 = (x1 - mean) * rs * g1[lane + 32] + be1[lane + 32];
    R1[q][lane] = y0;  R1[q][lane + 32] = y1;
    L1H[q][lane] = (_Float16)y0;  L1H[q][lane + 32] = (_Float16)y1;
  }
  __syncthreads();

  // FFN1: relu(ln1 @ Wf1 + bf1), N=1024 -> 64 col tiles, K=64 -> 2 chunks
#pragma unroll 1
  for (int ct = 0; ct < 64; ++ct) {
    v8f acc = {};
#pragma unroll
    for (int kt = 0; kt < 2; ++kt) {
      v16h a = lda_row(&L1H[ln][0], half, 32 * kt);
      v16h bm = ldb_packed(pw + PW_F1 + (size_t)(ct * 2 + kt) * 512, lane);
      acc = wmma_f16(a, bm, acc);
    }
    const float bias = bf1[ct * 16 + ln];
#pragma unroll
    for (int r = 0; r < 8; ++r)
      FFH[r + 8 * half][ct * 16 + ln] = (_Float16)fmaxf(acc[r] + bias, 0.0f);
  }
  __syncthreads();

  // FFN2: hidden @ Wf2 + bf2 + ln1 residual, K=1024 -> 32 chunks
#pragma unroll 1
  for (int c4 = 0; c4 < 4; ++c4) {
    v8f acc = {};
#pragma unroll 1
    for (int kt = 0; kt < 32; ++kt) {
      if (kt + 1 < 32)
        __builtin_prefetch(pw + PW_F2 + (size_t)(c4 * 32 + kt + 1) * 512 + lane * 16, 0, 1);
      v16h a = lda_row(&FFH[ln][0], half, 32 * kt);
      v16h bm = ldb_packed(pw + PW_F2 + (size_t)(c4 * 32 + kt) * 512, lane);
      acc = wmma_f16(a, bm, acc);
    }
    const float bias = bf2[c4 * 16 + ln];
#pragma unroll
    for (int r = 0; r < 8; ++r)
      R2[r + 8 * half][c4 * 16 + ln] =
          acc[r] + bias + R1[r + 8 * half][c4 * 16 + ln];
  }
  __syncthreads();

  // LN2 + transposed store back to [b, c, n, l]
#pragma unroll 1
  for (int q = 0; q < 16; ++q) {
    float x0 = R2[q][lane], x1 = R2[q][lane + 32];
    float mean = wsum(x0 + x1) * (1.0f / 64.0f);
    float var  = wsum(x0 * x0 + x1 * x1) * (1.0f / 64.0f) - mean * mean;
    float rs = rsqrtf(var + 1e-5f);
    float y0 = (x0 - mean) * rs * g2[lane]      + be2[lane];
    float y1 = (x1 - mean) * rs * g2[lane + 32] + be2[lane + 32];
    out[((size_t)(b * CCH + lane)      * NQ + n0 + q) * LQ + l] = y0;
    out[((size_t)(b * CCH + lane + 32) * NQ + n0 + q) * LQ + l] = y1;
  }
}

// ---------------- launcher ----------------
extern "C" void kernel_launch(void* const* d_in, const int* in_sizes, int n_in,
                              void* d_out, int out_size, void* d_ws, size_t ws_size,
                              hipStream_t stream) {
  (void)in_sizes; (void)n_in; (void)out_size;
  const float* x   = (const float*)d_in[0];
  const float* xa1 = (const float*)d_in[1];
  const float* xa2 = (const float*)d_in[2];
  const float* WQ1 = (const float*)d_in[3];  const float* bQ1 = (const float*)d_in[4];
  const float* WQ2 = (const float*)d_in[5];  const float* bQ2 = (const float*)d_in[6];
  const float* WK1 = (const float*)d_in[7];  const float* bK1 = (const float*)d_in[8];
  const float* WK2 = (const float*)d_in[9];  const float* bK2 = (const float*)d_in[10];
  const float* WV  = (const float*)d_in[11]; const float* bV  = (const float*)d_in[12];
  const float* Wo  = (const float*)d_in[13]; const float* bo  = (const float*)d_in[14];
  const float* Wf1 = (const float*)d_in[15]; const float* bf1 = (const float*)d_in[16];
  const float* Wf2 = (const float*)d_in[17]; const float* bf2 = (const float*)d_in[18];
  const float* g1  = (const float*)d_in[19]; const float* be1 = (const float*)d_in[20];
  const float* g2  = (const float*)d_in[21]; const float* be2 = (const float*)d_in[22];
  const float* temperature = (const float*)d_in[23];
  const float* alphas      = (const float*)d_in[24];

  // workspace layout (all f16)
  const size_t pw_elems  = PW_TOT;                          // 180224
  const size_t qc_elems  = (size_t)BL * NH * NP * 32;       // 2,555,904
  const size_t kc_elems  = qc_elems;
  const size_t vbp_elems = (size_t)BL * NH * KT * 512;      // 1,277,952
  const size_t aoh_elems = (size_t)BL * NQ * DM;            // 1,228,800
  const size_t need =
      (pw_elems + qc_elems + kc_elems + vbp_elems + aoh_elems) * sizeof(_Float16);
  if (ws_size < need) return;

  _Float16* pw  = (_Float16*)d_ws;
  _Float16* qc  = pw + pw_elems;
  _Float16* kc  = qc + qc_elems;
  _Float16* vbp = kc + kc_elems;
  _Float16* aoh = vbp + vbp_elems;

  float* out      = (float*)d_out;
  float* out_attn = out + (size_t)BSZ * CCH * NQ * LQ;            // 614400
  float* out_p2   = out_attn + (size_t)BSZ * NH * LQ * NQ * NQ;   // +30.72M

  // pack weights (B-tile f16 layout); zero V staging pads
  pack_w_kernel<<<dim3(32), 256, 0, stream>>>(WQ1, pw + PW_Q1, DM, CCH);
  pack_w_kernel<<<dim3(32), 256, 0, stream>>>(WQ2, pw + PW_Q2, DM, CCH);
  pack_w_kernel<<<dim3(32), 256, 0, stream>>>(WK1, pw + PW_K1, DM, CCH);
  pack_w_kernel<<<dim3(32), 256, 0, stream>>>(WK2, pw + PW_K2, DM, CCH);
  pack_w_kernel<<<dim3(32), 256, 0, stream>>>(WV,  pw + PW_V,  DM, CCH);
  pack_w_kernel<<<dim3(32), 256, 0, stream>>>(Wo,  pw + PW_O,  CCH, DM);
  pack_w_kernel<<<dim3(256), 256, 0, stream>>>(Wf1, pw + PW_F1, FFD, CCH);
  pack_w_kernel<<<dim3(256), 256, 0, stream>>>(Wf2, pw + PW_F2, CCH, FFD);
  zero_f16_kernel<<<dim3((unsigned)((vbp_elems + 255) / 256)), 256, 0, stream>>>(
      vbp, (long)vbp_elems);

  proj_kernel<<<dim3(NT, BL), 128, 0, stream>>>(
      x, xa1, xa2, pw, bQ1, bQ2, bK1, bK2, bV, qc, kc, vbp);

  attn_kernel<<<dim3(NT, BL, NH), 32, 0, stream>>>(
      qc, kc, vbp, temperature, alphas, out_attn, out_p2, aoh);

  epi_kernel<<<dim3(NT, BL), 32, 0, stream>>>(
      x, aoh, pw, bo, bf1, bf2, g1, be1, g2, be2, out);
}
